// PITLossEfficient_17386027614819
// MI455X (gfx1250) — matrix operand order repaired and checked by reading, not compiled, not run on hardware
//
#include <hip/hip_runtime.h>
#include <hip/hip_bf16.h>

typedef __attribute__((ext_vector_type(16))) _Float16 v16h;
typedef __attribute__((ext_vector_type(8)))  _Float16 v8h;
typedef __attribute__((ext_vector_type(8)))  float    v8f;

#define LN2f 0.69314718055994530942f
#define EPSf 1e-7f

__device__ __forceinline__ float clipf(float x) {
    const float HI = 1.0f - EPSf;            // rounds to 0.99999988f, matches jnp.clip in f32
    return fminf(fmaxf(x, EPSf), HI);
}

// Grid: B*Kt blocks of 256 threads (8 wave32s). Block (b,kt) streams a 4096-t
// slice of batch b. Each wave owns a 512-t subrange = 16 chunks of 32 t.
// Per chunk: coalesced float4 loads, logs, stage f16 g/target tiles to LDS in
// WMMA fragment order, one v_wmma_f32_16x16x32_f16 accumulating G (K-axis = t).
__global__ __launch_bounds__(256) void pit_accum_kernel(
    const float* __restrict__ pred, const float* __restrict__ target,
    const float* __restrict__ mask, float* __restrict__ partials,
    int T, int Kt, int tPerBlock)
{
    __shared__ __align__(16) _Float16 Ag[8][4][32];   // per-wave: g  tile, [i][t]
    __shared__ __align__(16) _Float16 Bt[8][4][32];   // per-wave: tgt tile, [j][t]
    __shared__ float sred[8 * 24];

    const int tid  = threadIdx.x;
    const int w    = tid >> 5;
    const int lane = tid & 31;
    const int b    = blockIdx.x / Kt;
    const int kt   = blockIdx.x % Kt;
    const int tPerWave = tPerBlock >> 3;              // 512
    const int nChunk   = tPerWave >> 5;               // 16
    const int tBase    = kt * tPerBlock + w * tPerWave;

    const size_t rowBase = (size_t)b * (size_t)T;

    v8f c;
    #pragma unroll
    for (int i = 0; i < 8; ++i) c[i] = 0.0f;
    float a0 = 0.f, a1 = 0.f, a2 = 0.f, a3 = 0.f, msum = 0.f;

    const int ii = (lane & 15) & 3;                   // replicated row index i = m%4
    const v8h* pa = (const v8h*)(&Ag[w][ii][0]);      // 16B-aligned 8-half views
    const v8h* pb = (const v8h*)(&Bt[w][ii][0]);
    const int hiHalf = lane >> 4;                     // 0: lanes 0-15, 1: lanes 16-31

    for (int ch = 0; ch < nChunk; ++ch) {
        const int t = tBase + ch * 32 + lane;
        const size_t e = (rowBase + (size_t)t) * 4;
        const float4 pr = *(const float4*)(pred + e);
        const float4 tg = *(const float4*)(target + e);
        const float  mk = mask[rowBase + (size_t)t];
        __builtin_prefetch(pred + e + 128, 0, 0);     // next chunk, streaming hint
        __builtin_prefetch(target + e + 128, 0, 0);

        const float p0 = clipf(pr.x), p1 = clipf(pr.y), p2 = clipf(pr.z), p3 = clipf(pr.w);
        const float lp0 = __log2f(p0) * LN2f, lp1 = __log2f(p1) * LN2f;
        const float lp2 = __log2f(p2) * LN2f, lp3 = __log2f(p3) * LN2f;
        const float l10 = __log2f(1.0f - p0) * LN2f, l11 = __log2f(1.0f - p1) * LN2f;
        const float l12 = __log2f(1.0f - p2) * LN2f, l13 = __log2f(1.0f - p3) * LN2f;

        a0 += mk * l10; a1 += mk * l11; a2 += mk * l12; a3 += mk * l13;
        msum += mk;

        // stage A (g, masked) and B (target) tiles, [i][t] so fragment reads are b128
        Ag[w][0][lane] = (_Float16)(mk * (lp0 - l10));
        Ag[w][1][lane] = (_Float16)(mk * (lp1 - l11));
        Ag[w][2][lane] = (_Float16)(mk * (lp2 - l12));
        Ag[w][3][lane] = (_Float16)(mk * (lp3 - l13));
        Bt[w][0][lane] = (_Float16)tg.x;
        Bt[w][1][lane] = (_Float16)tg.y;
        Bt[w][2][lane] = (_Float16)tg.z;
        Bt[w][3][lane] = (_Float16)tg.w;
        // same-wave LDS RAW: DS ops are in-order per wave; compiler inserts dscnt waits

        // A 16x32 f16 layout: lanes 0-15 hold K=0-7 (halves 0-7) and K=16-23 (8-15);
        // lanes 16-31 hold K=8-15 and K=24-31.  K == t within chunk.
        const v8h aLo = pa[hiHalf];
        const v8h aHi = pa[2 + hiHalf];
        // B 32x16 f16 layout: lanes 0-15 hold K=0-7 and K=16-23; lanes 16-31 hold K=8-15 and K=24-31.
        const v8h bLo = pb[hiHalf];
        const v8h bHi = pb[2 + hiHalf];
        v16h av, bv;
        #pragma unroll
        for (int h = 0; h < 8; ++h) {
            av[h] = aLo[h]; av[h + 8] = aHi[h];
            bv[h] = bLo[h]; bv[h + 8] = bHi[h];
        }
        c = __builtin_amdgcn_wmma_f32_16x16x32_f16(
                false, av, false, bv, (short)0, c, false, false);
    }

    // wave-level reduce of scalar partials
    #pragma unroll
    for (int off = 16; off > 0; off >>= 1) {
        a0 += __shfl_down(a0, off, 32);
        a1 += __shfl_down(a1, off, 32);
        a2 += __shfl_down(a2, off, 32);
        a3 += __shfl_down(a3, off, 32);
        msum += __shfl_down(msum, off, 32);
    }
    float* my = &sred[w * 24];
    if (lane < 4) {
        // D layout: VGPR r, lanes 0-15 -> M=r, N=lane. G[i][j] = c[i] at lane j.
        my[0 * 4 + lane] = c[0];
        my[1 * 4 + lane] = c[1];
        my[2 * 4 + lane] = c[2];
        my[3 * 4 + lane] = c[3];
    }
    if (lane == 0) { my[16] = a0; my[17] = a1; my[18] = a2; my[19] = a3; my[20] = msum; }
    __syncthreads();
    if (tid < 21) {
        float s = 0.f;
        #pragma unroll
        for (int ww = 0; ww < 8; ++ww) s += sred[ww * 24 + tid];
        partials[(size_t)blockIdx.x * 24 + tid] = s;   // fully overwritten: no ws zeroing needed
    }
}

// One block, thread b per batch: sum t-splits in fixed order (deterministic),
// build 4x4 cost, brute-force 24 permutations, average over batches.
__global__ __launch_bounds__(128) void pit_final_kernel(
    const float* __restrict__ partials, float* __restrict__ out, int Kt, int B)
{
    const int b = threadIdx.x;
    float G[16], a[4], msum = 0.f;
    #pragma unroll
    for (int k = 0; k < 16; ++k) G[k] = 0.f;
    #pragma unroll
    for (int i = 0; i < 4; ++i) a[i] = 0.f;

    float bl = 0.f;
    if (b < B) {
        for (int kt = 0; kt < Kt; ++kt) {
            const float* p = partials + (size_t)(b * Kt + kt) * 24;
            #pragma unroll
            for (int k = 0; k < 16; ++k) G[k] += p[k];
            #pragma unroll
            for (int i = 0; i < 4; ++i) a[i] += p[16 + i];
            msum += p[20];
        }
        const float inv = 1.0f / msum;
        float cost[4][4];
        #pragma unroll
        for (int i = 0; i < 4; ++i)
            #pragma unroll
            for (int j = 0; j < 4; ++j)
                cost[i][j] = -(a[i] + G[i * 4 + j]) * inv;

        float best = 3.402823466e38f;
        #pragma unroll
        for (int q0 = 0; q0 < 4; ++q0)
            #pragma unroll
            for (int q1 = 0; q1 < 4; ++q1) {
                if (q1 == q0) continue;
                #pragma unroll
                for (int q2 = 0; q2 < 4; ++q2) {
                    if (q2 == q0 || q2 == q1) continue;
                    const int q3 = 6 - q0 - q1 - q2;
                    const float tot = cost[0][q0] + cost[1][q1] + cost[2][q2] + cost[3][q3];
                    best = fminf(best, tot);
                }
            }
        bl = best * 0.25f;                             // mean over S=4 sources
    }

    #pragma unroll
    for (int off = 16; off > 0; off >>= 1) bl += __shfl_down(bl, off, 32);
    __shared__ float r[4];
    if ((threadIdx.x & 31) == 0) r[threadIdx.x >> 5] = bl;
    __syncthreads();
    if (threadIdx.x == 0) out[0] = (r[0] + r[1] + r[2] + r[3]) / (float)B;
}

extern "C" void kernel_launch(void* const* d_in, const int* in_sizes, int n_in,
                              void* d_out, int out_size, void* d_ws, size_t ws_size,
                              hipStream_t stream) {
    const float* pred   = (const float*)d_in[0];
    const float* target = (const float*)d_in[1];
    const float* mask   = (const float*)d_in[2];
    float* out = (float*)d_out;

    const int B = 128;                    // reference setup: (B,T,S)=(128,32768,4)
    const int T = in_sizes[2] / B;        // mask is B*T
    const int tPerBlock = 4096;           // 8 waves x 16 chunks x 32 t
    const int Kt = T / tPerBlock;         // 8

    float* partials = (float*)d_ws;       // B*Kt*24 floats = 98 KB scratch

    hipLaunchKernelGGL(pit_accum_kernel, dim3(B * Kt), dim3(256), 0, stream,
                       pred, target, mask, partials, T, Kt, tPerBlock);
    hipLaunchKernelGGL(pit_final_kernel, dim3(1), dim3(128), 0, stream,
                       partials, out, Kt, B);
}
